// GPTModel_51110110822883
// MI455X (gfx1250) — compile-verified, matrix operand order checked by reading, not run on hardware
//
#include <hip/hip_runtime.h>
#include <hip/hip_bf16.h>

// ---------------------------------------------------------------------------
// GPT-2-ish forward for MI455X (gfx1250): bf16 WMMA + TDM-staged GEMM tiles.
// Model constants (from reference): V=32000 S=1024 D=768 L=6 H=12 FF=3072
// ---------------------------------------------------------------------------

typedef __attribute__((ext_vector_type(16))) __bf16 v16bf;
typedef __attribute__((ext_vector_type(8)))  float  v8f;
typedef __attribute__((ext_vector_type(4)))  unsigned int u32x4;
typedef __attribute__((ext_vector_type(8)))  int  i32x8;
typedef __attribute__((ext_vector_type(4)))  int  i32x4;

#if defined(__has_builtin)
#if __has_builtin(__builtin_amdgcn_tensor_load_to_lds) && \
    __has_builtin(__builtin_amdgcn_s_wait_tensorcnt)
#define HAVE_TDM 1
#endif
#endif
#ifndef HAVE_TDM
#define HAVE_TDM 0
#endif

union FragU { v16bf v; unsigned int u[8]; };

__device__ __forceinline__ unsigned short f2bf(float f) {
  unsigned int u = __float_as_uint(f);
  u += 0x7FFFu + ((u >> 16) & 1u);      // round-to-nearest-even
  return (unsigned short)(u >> 16);
}

__device__ __forceinline__ v8f wmma_bf16(v16bf a, v16bf b, v8f c) {
  // v_wmma_f32_16x16x32_bf16  D = A*B + C   (fp32 accumulate)
  return __builtin_amdgcn_wmma_f32_16x16x32_bf16(
      /*neg_a=*/false, a, /*neg_b=*/false, b,
      /*c_mod=*/(short)0, c, /*reuse_a=*/false, /*reuse_b=*/false);
}

// A-style fragment (16 rows x 32 K) from row-major bf16 [rows][ld], K contiguous.
// ISA 7.12.2: lane holds row = lane&15; lanes>=16 take the upper K-half of each group.
__device__ __forceinline__ v16bf load_frag_k(const unsigned short* base,
                                             int ld, int row0, int k0) {
  const int lane = threadIdx.x & 31;
  const int row  = row0 + (lane & 15);
  const int half = lane >> 4;
  const unsigned short* p = base + (size_t)row * ld + k0 + half * 8;
  FragU f;
#pragma unroll
  for (int i = 0; i < 8; ++i) {
    const int k = (i >> 2) * 16 + (i & 3) * 2;     // two consecutive bf16 per VGPR
    f.u[i] = *(const unsigned int*)(p + k);        // 4B-aligned (ld,k0 even)
  }
  return f.v;
}

// B fragment built from the *columns* of a K-major matrix (e.g. V tile [32k x 16d]).
__device__ __forceinline__ v16bf load_frag_col(const unsigned short* __restrict__ base,
                                               int ld, int k0, int col0) {
  const int lane = threadIdx.x & 31;
  const int col  = col0 + (lane & 15);
  const int half = lane >> 4;
  const unsigned short* p = base + (size_t)(k0 + half * 8) * ld + col;
  FragU f;
#pragma unroll
  for (int i = 0; i < 8; ++i) {
    const int k = (i >> 2) * 16 + (i & 3) * 2;
    unsigned int lo = p[(size_t)k * ld];
    unsigned int hi = p[(size_t)(k + 1) * ld];
    f.u[i] = lo | (hi << 16);
  }
  return f.v;
}

// ---------------------------------------------------------------------------
// Tensor Data Mover: stage a [tileRows x tileK] bf16 tile (row stride tensK)
// into LDS at byte offset lds_off.  Builds a real 2D D# per ISA ch.8:
// group0 = {count=1 | lds_addr | global_addr | type=2}, group1 = {data_size=2B,
// tensor_dim0/1, tile_dim0/1, tensor_dim0_stride}.  Groups 2/3 unused (2D).
// ---------------------------------------------------------------------------
#if HAVE_TDM
__device__ __forceinline__ void tdm_load_tile(const unsigned short* gsrc,
                                              unsigned lds_off, int tensK,
                                              int tensN, int tileRows, int tileK) {
  unsigned long long ga = (unsigned long long)(size_t)gsrc;
  u32x4 g0;
  g0[0] = 1u;                                            // count=1 (valid), user mode
  g0[1] = lds_off;                                       // LDS byte address
  g0[2] = (unsigned)(ga & 0xFFFFFFFFu);                  // global_addr[31:0]
  g0[3] = (unsigned)((ga >> 32) & 0x01FFFFFFu) | (2u << 30); // addr[56:32] | type=2
  i32x8 g1;
  g1[0] = (int)(1u << 16);                               // wg_mask=0, data_size=1 (2B)
  g1[1] = (int)(((unsigned)tensK & 0xFFFFu) << 16);      // tensor_dim0[15:0]
  g1[2] = (int)((((unsigned)tensK >> 16) & 0xFFFFu) |
                (((unsigned)tensN & 0xFFFFu) << 16));    // dim0 hi | tensor_dim1 lo
  g1[3] = (int)((((unsigned)tensN >> 16) & 0xFFFFu) |
                (((unsigned)tileK & 0xFFFFu) << 16));    // dim1 hi | tile_dim0
  g1[4] = (int)((unsigned)tileRows & 0xFFFFu);           // tile_dim1 (tile_dim2=0)
  g1[5] = tensK;                                         // tensor_dim0_stride[31:0]
  g1[6] = 0;                                             // stride hi / dim1_stride lo
  g1[7] = 0;
  i32x4 z4 = {0, 0, 0, 0};
#if __clang_major__ >= 23
  i32x8 z8 = {0, 0, 0, 0, 0, 0, 0, 0};
  __builtin_amdgcn_tensor_load_to_lds(g0, g1, z4, z4, z8, 0);
#else
  __builtin_amdgcn_tensor_load_to_lds(g0, g1, z4, z4, 0);
#endif
}
#endif

// ---------------------------------------------------------------------------
// fp32 -> bf16 bulk convert (weights; one-time, then L2-resident)
// ---------------------------------------------------------------------------
__global__ void cvt_f32_bf16(const float* __restrict__ in,
                             unsigned short* __restrict__ out, long n) {
  long i = (long)blockIdx.x * blockDim.x + threadIdx.x;
  if (i < n) out[i] = f2bf(in[i]);
}

// ---------------------------------------------------------------------------
// h = tok_emb[idx] + pos_emb   (fp32 residual stream)
// ---------------------------------------------------------------------------
__global__ void embed_kernel(const int* __restrict__ idx,
                             const float* __restrict__ tok,
                             const float* __restrict__ pos,
                             float* __restrict__ h, int total, int T, int Dm) {
  int i = blockIdx.x * blockDim.x + threadIdx.x;
  if (i >= total) return;
  int d = i % Dm;
  int bt = i / Dm;
  int t = bt % T;
  h[i] = tok[(size_t)idx[bt] * Dm + d] + pos[(size_t)t * Dm + d];
}

// ---------------------------------------------------------------------------
// LayerNorm (fp32 stats) -> bf16 activation.  One 256-thread block per row.
// ---------------------------------------------------------------------------
__global__ void __launch_bounds__(256)
layernorm_bf16(const float* __restrict__ X, const float* __restrict__ W,
               const float* __restrict__ Bi, unsigned short* __restrict__ Y, int Dm) {
  const int row = blockIdx.x;
  const float* x = X + (size_t)row * Dm;
  __shared__ float red[256];
  float s = 0.f;
  for (int d = threadIdx.x; d < Dm; d += 256) s += x[d];
  red[threadIdx.x] = s; __syncthreads();
  for (int o = 128; o > 0; o >>= 1) {
    if (threadIdx.x < o) red[threadIdx.x] += red[threadIdx.x + o];
    __syncthreads();
  }
  const float mu = red[0] / Dm;
  __syncthreads();
  float v = 0.f;
  for (int d = threadIdx.x; d < Dm; d += 256) { float t = x[d] - mu; v += t * t; }
  red[threadIdx.x] = v; __syncthreads();
  for (int o = 128; o > 0; o >>= 1) {
    if (threadIdx.x < o) red[threadIdx.x] += red[threadIdx.x + o];
    __syncthreads();
  }
  const float rs = rsqrtf(red[0] / Dm + 1e-5f);
  for (int d = threadIdx.x; d < Dm; d += 256)
    Y[(size_t)row * Dm + d] = f2bf((x[d] - mu) * rs * W[d] + Bi[d]);
}

// ---------------------------------------------------------------------------
// C[M,N] = A[M,K](bf16) @ W[N,K]^T(bf16)
// Block = 128 threads (4 waves) computing a 128(M) x 64(N) tile.
// The 64x32 weight tile is TDM-staged into LDS (double buffered, TENSORcnt
// synchronized) and shared by all 4 waves; each wave computes 32x64 with
// 8 chained v_wmma_f32_16x16x32_bf16 per K-step (2 A frags x 4 B frags).
// EPI: 0 = bf16 store, 1 = fp32 store + residual, 2 = exact GELU -> bf16,
//      3 = fp32 store.
// ---------------------------------------------------------------------------
template <int EPI>
__global__ void __launch_bounds__(128)
gemm128x64(const unsigned short* __restrict__ A, const unsigned short* __restrict__ W,
           float* Cf, unsigned short* __restrict__ Cb, const float* Res,
           int M, int N, int K) {
  constexpr unsigned TILE_BYTES = 64 * 32 * 2;           // 4 KB per buffer
  __shared__ __align__(16) unsigned short Wt[2][64 * 32];
  const int n0 = blockIdx.x * 64;
  const int m0 = blockIdx.y * 128 + (threadIdx.x >> 5) * 32;
  const int lane = threadIdx.x & 31;
  v8f c[2][4] = {};
  int parity = 0;

#if HAVE_TDM
  if (threadIdx.x < 32)                                   // wave 0 drives the TDM
    tdm_load_tile(W + (size_t)n0 * K, 0u, K, N, 64, 32);
#endif

  for (int k0 = 0; k0 < K; k0 += 32) {
#if HAVE_TDM
    const bool more = (k0 + 32) < K;
    if (threadIdx.x < 32) {
      if (more)
        tdm_load_tile(W + (size_t)n0 * K + (k0 + 32),
                      (unsigned)(parity ^ 1) * TILE_BYTES, K, N, 64, 32);
      if (more) __builtin_amdgcn_s_wait_tensorcnt(1);     // current buffer done
      else      __builtin_amdgcn_s_wait_tensorcnt(0);
    }
    __syncthreads();
    const unsigned short* wt = &Wt[parity][0];
#else
    __syncthreads();
    {
      const unsigned short* src = W + (size_t)n0 * K + k0;
      unsigned int* dst = (unsigned int*)&Wt[0][0];
      for (int i = threadIdx.x; i < 64 * 16; i += 128) {
        const int r = i >> 4, kk = (i & 15) * 2;
        dst[i] = *(const unsigned int*)(src + (size_t)r * K + kk);
      }
    }
    __syncthreads();
    const unsigned short* wt = &Wt[0][0];
#endif
    const v16bf a0 = load_frag_k(A, K, m0, k0);
    const v16bf a1 = load_frag_k(A, K, m0 + 16, k0);
#pragma unroll
    for (int j = 0; j < 4; ++j) {
      const v16bf b = load_frag_k(wt, 32, j * 16, 0);     // ds_load from LDS
      c[0][j] = wmma_bf16(a0, b, c[0][j]);
      c[1][j] = wmma_bf16(a1, b, c[1][j]);
    }
    __syncthreads();                                      // before buffer reuse
    parity ^= 1;
  }

  const int half = lane >> 4, nl = lane & 15;
#pragma unroll
  for (int i = 0; i < 2; ++i) {
#pragma unroll
    for (int j = 0; j < 4; ++j) {
#pragma unroll
      for (int r = 0; r < 8; ++r) {
        const int m = m0 + i * 16 + 8 * half + r;
        const int n = n0 + j * 16 + nl;
        const size_t off = (size_t)m * N + n;
        const float v = c[i][j][r];
        if (EPI == 0) {
          Cb[off] = f2bf(v);
        } else if (EPI == 1) {
          Cf[off] = v + Res[off];
        } else if (EPI == 2) {
          Cb[off] = f2bf(0.5f * v * (1.0f + erff(v * 0.70710678f)));  // exact GELU
        } else {
          Cf[off] = v;
        }
      }
    }
  }
}

// ---------------------------------------------------------------------------
// Flash-style causal attention. One wave per (batch, head, 16-row q tile).
// S = Q K^T via chained 16x16x32 bf16 WMMA; online softmax with wave32
// shfl_xor row reductions; P routed C-layout -> A-layout through LDS; O += P V.
// qkv layout: [B, T, 3*768] bf16 (q | k | v), head h at column h*64.
// ---------------------------------------------------------------------------
__global__ void __launch_bounds__(32)
attn_kernel(const unsigned short* __restrict__ qkv, unsigned short* __restrict__ Y) {
  const int qt = blockIdx.x, hh = blockIdx.y, b = blockIdx.z;
  const int T = 1024, Dm = 768, ldq = 3 * Dm;
  const int m0 = qt * 16;
  const int lane = threadIdx.x;
  const int half = lane >> 4, nl = lane & 15;

  const unsigned short* Qb = qkv + (size_t)b * T * ldq + hh * 64;
  const unsigned short* Kb = Qb + Dm;
  const unsigned short* Vb = Qb + 2 * Dm;

  __shared__ __align__(16) unsigned short Plds[16 * 32];

  const v16bf qf0 = load_frag_k(Qb, ldq, m0, 0);
  const v16bf qf1 = load_frag_k(Qb, ldq, m0, 32);

  v8f o[4] = {};
  float rowM[8], rowL[8];
#pragma unroll
  for (int r = 0; r < 8; ++r) { rowM[r] = -1e30f; rowL[r] = 0.f; }

  const float scale = 0.125f;  // 1/sqrt(64)
  const int kbmax = (m0 + 15) / 32;
  for (int kb = 0; kb <= kbmax; ++kb) {
    const int kbase = kb * 32;
    v8f s0 = {}, s1 = {};
    s0 = wmma_bf16(qf0, load_frag_k(Kb, ldq, kbase, 0), s0);
    s0 = wmma_bf16(qf1, load_frag_k(Kb, ldq, kbase, 32), s0);
    s1 = wmma_bf16(qf0, load_frag_k(Kb, ldq, kbase + 16, 0), s1);
    s1 = wmma_bf16(qf1, load_frag_k(Kb, ldq, kbase + 16, 32), s1);

    const int key0 = kbase + nl, key1 = kbase + 16 + nl;
#pragma unroll
    for (int r = 0; r < 8; ++r) {
      const int q = m0 + 8 * half + r;
      float v0 = (key0 <= q) ? s0[r] * scale : -1e30f;
      float v1 = (key1 <= q) ? s1[r] * scale : -1e30f;
      float mloc = fmaxf(v0, v1);
#pragma unroll
      for (int msk = 8; msk >= 1; msk >>= 1)
        mloc = fmaxf(mloc, __shfl_xor(mloc, msk, 32));
      const float mnew = fmaxf(rowM[r], mloc);
      const float alpha = __expf(rowM[r] - mnew);
      rowM[r] = mnew;
      const float p0 = __expf(v0 - mnew);
      const float p1 = __expf(v1 - mnew);
      float ls = p0 + p1;
#pragma unroll
      for (int msk = 8; msk >= 1; msk >>= 1) ls += __shfl_xor(ls, msk, 32);
      rowL[r] = rowL[r] * alpha + ls;
#pragma unroll
      for (int j = 0; j < 4; ++j) o[j][r] *= alpha;
      const int rr = 8 * half + r;
      Plds[rr * 32 + nl]      = f2bf(p0);
      Plds[rr * 32 + 16 + nl] = f2bf(p1);
    }
    __syncthreads();
    const v16bf pf = load_frag_k(Plds, 32, 0, 0);   // P as 16x32 A fragment
#pragma unroll
    for (int j = 0; j < 4; ++j)
      o[j] = wmma_bf16(pf, load_frag_col(Vb, ldq, kbase, j * 16), o[j]);
    __syncthreads();
  }

#pragma unroll
  for (int r = 0; r < 8; ++r) {
    const float inv = 1.0f / rowL[r];
    const int m = m0 + 8 * half + r;
#pragma unroll
    for (int j = 0; j < 4; ++j)
      Y[((size_t)b * T + m) * Dm + hh * 64 + j * 16 + nl] = f2bf(o[j][r] * inv);
  }
}

// ---------------------------------------------------------------------------
// Host orchestration
// ---------------------------------------------------------------------------
extern "C" void kernel_launch(void* const* d_in, const int* in_sizes, int n_in,
                              void* d_out, int out_size, void* d_ws, size_t ws_size,
                              hipStream_t stream) {
  (void)in_sizes; (void)n_in; (void)out_size; (void)ws_size;
  const int B = 4, T = 1024, Dm = 768, Lc = 6, FFc = 3072, Vv = 32000;
  const int BT = B * T;                    // 4096 rows; all dims multiples of 64/128

  const int*   idx  = (const int*)  d_in[0];
  const float* tok  = (const float*)d_in[1];
  const float* pos  = (const float*)d_in[2];
  const float* ln1w = (const float*)d_in[3];
  const float* ln1b = (const float*)d_in[4];
  const float* qkvw = (const float*)d_in[5];
  const float* outw = (const float*)d_in[6];
  const float* ln2w = (const float*)d_in[7];
  const float* ln2b = (const float*)d_in[8];
  const float* upw  = (const float*)d_in[9];
  const float* dww  = (const float*)d_in[10];
  const float* lnfw = (const float*)d_in[11];
  const float* lnfb = (const float*)d_in[12];
  float* logits = (float*)d_out;

  // workspace carve-up
  char* wsp = (char*)d_ws;
  size_t off = 0;
  auto carve = [&](size_t bytes) -> void* {
    void* p = wsp + off;
    off += (bytes + 255) & ~(size_t)255;
    return p;
  };
  const size_t nWq = (size_t)Lc * 3 * Dm * Dm;
  const size_t nWo = (size_t)Lc * Dm * Dm;
  const size_t nWu = (size_t)Lc * FFc * Dm;
  const size_t nWd = (size_t)Lc * Dm * FFc;
  const size_t nTe = (size_t)Vv * Dm;
  unsigned short* wq   = (unsigned short*)carve(nWq * 2);
  unsigned short* wo   = (unsigned short*)carve(nWo * 2);
  unsigned short* wu   = (unsigned short*)carve(nWu * 2);
  unsigned short* wd   = (unsigned short*)carve(nWd * 2);
  unsigned short* temb = (unsigned short*)carve(nTe * 2);
  float*          hbuf = (float*)         carve((size_t)BT * Dm * 4);
  unsigned short* xbf  = (unsigned short*)carve((size_t)BT * Dm * 2);
  unsigned short* qkvb = (unsigned short*)carve((size_t)BT * 3 * Dm * 2);
  unsigned short* ybf  = (unsigned short*)carve((size_t)BT * Dm * 2);
  unsigned short* ubf  = (unsigned short*)carve((size_t)BT * FFc * 2);

  auto cvt = [&](const float* src, unsigned short* dst, size_t n) {
    cvt_f32_bf16<<<dim3((unsigned)((n + 255) / 256)), 256, 0, stream>>>(src, dst, (long)n);
  };
  // one-time bf16 weight conversion (whole model then fits in 192MB L2)
  cvt(qkvw, wq, nWq);
  cvt(outw, wo, nWo);
  cvt(upw,  wu, nWu);
  cvt(dww,  wd, nWd);
  cvt(tok,  temb, nTe);

  // embeddings
  {
    int total = BT * Dm;
    embed_kernel<<<dim3((total + 255) / 256), 256, 0, stream>>>(idx, tok, pos, hbuf,
                                                                total, T, Dm);
  }

  const dim3 blk(128);
  for (int l = 0; l < Lc; ++l) {
    // x = LN1(h) -> bf16
    layernorm_bf16<<<dim3(BT), 256, 0, stream>>>(hbuf, ln1w + l * Dm, ln1b + l * Dm,
                                                 xbf, Dm);
    // qkv = x @ qkv_w^T  (bf16 out)
    gemm128x64<0><<<dim3(3 * Dm / 64, BT / 128), blk, 0, stream>>>(
        xbf, wq + (size_t)l * 3 * Dm * Dm, nullptr, qkvb, nullptr, BT, 3 * Dm, Dm);
    // attention -> y (bf16)
    attn_kernel<<<dim3(T / 16, 12, B), dim3(32), 0, stream>>>(qkvb, ybf);
    // h = h + y @ out_w^T
    gemm128x64<1><<<dim3(Dm / 64, BT / 128), blk, 0, stream>>>(
        ybf, wo + (size_t)l * Dm * Dm, hbuf, nullptr, hbuf, BT, Dm, Dm);
    // x = LN2(h)
    layernorm_bf16<<<dim3(BT), 256, 0, stream>>>(hbuf, ln2w + l * Dm, ln2b + l * Dm,
                                                 xbf, Dm);
    // u = gelu(x @ up_w^T)  (bf16)
    gemm128x64<2><<<dim3(FFc / 64, BT / 128), blk, 0, stream>>>(
        xbf, wu + (size_t)l * FFc * Dm, nullptr, ubf, nullptr, BT, FFc, Dm);
    // h = h + u @ down_w^T
    gemm128x64<1><<<dim3(Dm / 64, BT / 128), blk, 0, stream>>>(
        ubf, wd + (size_t)l * Dm * FFc, hbuf, nullptr, hbuf, BT, Dm, FFc);
  }

  // final LN + tied lm_head (dominant GEMM: 4096 x 32000 x 768)
  layernorm_bf16<<<dim3(BT), 256, 0, stream>>>(hbuf, lnfw, lnfb, xbf, Dm);
  gemm128x64<3><<<dim3(Vv / 64, BT / 128), blk, 0, stream>>>(
      xbf, temb, logits, nullptr, nullptr, BT, Vv, Dm);
}
